// DotProductAttention_35218731827957
// MI455X (gfx1250) — matrix operand order repaired and checked by reading, not compiled, not run on hardware
//
#include <hip/hip_runtime.h>
#include <stdint.h>

// ---------------------------------------------------------------------------
// DotProductAttention for MI455X (gfx1250), wave32 + v_wmma_f32_16x16x32_bf16
//
// Pipeline (all bf16 storage, f32 accumulate):
//   1. convert weights (transposed) and x to bf16
//   2. mlp_kernel  : q = MLP(x_target), k = MLP(x_context)    (WMMA)
//   3. qk_kernel   : S = q k^T / 16, stored bf16              (WMMA)
//   4. stats_kernel: per-column (axis=1) m_c, Z_c; S <- exp(S-m_c)
//   5. rscale      : r'T[d][c] = r[c][d] / Z_c   (LDS transpose)
//   6. pv_kernel   : out = E @ r'                              (WMMA)
// ---------------------------------------------------------------------------

typedef __bf16 bf16_t;
typedef bf16_t v16bf __attribute__((ext_vector_type(16)));
typedef float  v8f   __attribute__((ext_vector_type(8)));

#define NBATCH 16
#define NCTX   2048
#define NTGT   2048
#define IND    128
#define DDIM   256

__device__ __forceinline__ uint16_t f2bf(float f) {
  uint32_t u = __float_as_uint(f);
  u += 0x7FFFu + ((u >> 16) & 1u);      // round-to-nearest-even
  return (uint16_t)(u >> 16);
}
__device__ __forceinline__ float bf2f(uint16_t h) {
  return __uint_as_float(((uint32_t)h) << 16);
}

union FragU { uint32_t u[8]; v16bf v; };

// A fragment: 16x32 bf16 tile from row-major memory, base at (row0, k0), row
// stride ld elements.  ISA 16-bit A 16x32 layout:
//   lanes 0-15 : VGPR0-3 -> K=0..7,  VGPR4-7 -> K=16..23
//   lanes 16-31: VGPR0-3 -> K=8..15, VGPR4-7 -> K=24..31
__device__ __forceinline__ v16bf load_a(const uint16_t* base, int ld, int l16, int half) {
  FragU f;
#pragma unroll
  for (int j = 0; j < 8; ++j) {
    int kp = ((j & 4) << 2) + 8 * half + 2 * (j & 3);
    f.u[j] = *(const uint32_t*)(base + (size_t)l16 * ld + kp);
  }
  return f.v;
}

// B fragment: 32x16 bf16; memory holds B^T row-major (bt[n][k]), base at
// (n0, k0), row stride ld.  Same K order as A: lanes 0-15 VGPR0-3 K=0..7,
// VGPR4-7 K=16..23; lanes 16-31 VGPR0-3 K=8..15, VGPR4-7 K=24..31.
__device__ __forceinline__ v16bf load_bt(const uint16_t* base, int ld, int l16, int half) {
  FragU f;
#pragma unroll
  for (int j = 0; j < 8; ++j) {
    int kk = ((j & 4) << 2) + 8 * half + 2 * (j & 3);
    f.u[j] = *(const uint32_t*)(base + (size_t)l16 * ld + kk);
  }
  return f.v;
}

__device__ __forceinline__ v8f wmma_bf16(v16bf a, v16bf b, v8f c) {
  return __builtin_amdgcn_wmma_f32_16x16x32_bf16(false, a, false, b, (short)0, c,
                                                 false, false);
}

// ---------------------------------------------------------------------------
// dtype converts
// ---------------------------------------------------------------------------
__global__ __launch_bounds__(256) void cvt_bf16_kernel(const float* __restrict__ in,
                                                       uint16_t* __restrict__ out, int n) {
  int i = blockIdx.x * 256 + threadIdx.x;
  if (i < n) out[i] = f2bf(in[i]);
}

// w [K][N] f32 -> wT [N][K] bf16
__global__ __launch_bounds__(256) void cvt_wT_kernel(const float* __restrict__ w,
                                                     uint16_t* __restrict__ wT,
                                                     int K, int N) {
  int i = blockIdx.x * 256 + threadIdx.x;
  if (i < K * N) {
    int k = i % K, n = i / K;
    wT[i] = f2bf(w[(size_t)k * N + n]);
  }
}

// ---------------------------------------------------------------------------
// MLP: 128 -> 256 (relu) -> 256 (relu) -> 256.  2 waves/block, 16 rows/wave.
// ---------------------------------------------------------------------------
__global__ __launch_bounds__(64) void mlp_kernel(
    const uint16_t* __restrict__ x,                                  // [rows][128]
    const uint16_t* __restrict__ w1T, const float* __restrict__ b1,  // [256][128]
    const uint16_t* __restrict__ w2T, const float* __restrict__ b2,  // [256][256]
    const uint16_t* __restrict__ w3T, const float* __restrict__ b3,  // [256][256]
    uint16_t* __restrict__ y)                                        // [rows][256]
{
  __shared__ uint16_t h1[2][16 * 256];
  __shared__ uint16_t h2[2][16 * 256];
  const int tid  = threadIdx.x;
  const int wave = tid >> 5;
  const int lane = tid & 31;
  const int half = lane >> 4;
  const int l16  = lane & 15;
  const size_t row0 = (size_t)blockIdx.x * 32 + (size_t)wave * 16;
  const v8f vzero = {};

  // ---- layer 1: [16x128] @ w1 ----
  for (int nh = 0; nh < 2; ++nh) {
    const int n0 = nh * 128;
    v8f acc[8];
#pragma unroll
    for (int j = 0; j < 8; ++j) acc[j] = vzero;
    for (int kc = 0; kc < 4; ++kc) {
      const int k0 = kc * 32;
      v16bf a = load_a(x + row0 * IND + k0, IND, l16, half);
#pragma unroll
      for (int j = 0; j < 8; ++j) {
        v16bf b = load_bt(w1T + (size_t)(n0 + 16 * j) * IND + k0, IND, l16, half);
        acc[j] = wmma_bf16(a, b, acc[j]);
      }
    }
#pragma unroll
    for (int j = 0; j < 8; ++j) {
      float bb = b1[n0 + 16 * j + l16];
#pragma unroll
      for (int v = 0; v < 8; ++v) {
        float f = fmaxf(acc[j][v] + bb, 0.0f);
        h1[wave][(v + 8 * half) * 256 + n0 + 16 * j + l16] = f2bf(f);
      }
    }
  }
  __syncthreads();

  // ---- layer 2: [16x256] @ w2 ----
  for (int nh = 0; nh < 2; ++nh) {
    const int n0 = nh * 128;
    v8f acc[8];
#pragma unroll
    for (int j = 0; j < 8; ++j) acc[j] = vzero;
    for (int kc = 0; kc < 8; ++kc) {
      const int k0 = kc * 32;
      v16bf a = load_a(&h1[wave][k0], 256, l16, half);
#pragma unroll
      for (int j = 0; j < 8; ++j) {
        v16bf b = load_bt(w2T + (size_t)(n0 + 16 * j) * 256 + k0, 256, l16, half);
        acc[j] = wmma_bf16(a, b, acc[j]);
      }
    }
#pragma unroll
    for (int j = 0; j < 8; ++j) {
      float bb = b2[n0 + 16 * j + l16];
#pragma unroll
      for (int v = 0; v < 8; ++v) {
        float f = fmaxf(acc[j][v] + bb, 0.0f);
        h2[wave][(v + 8 * half) * 256 + n0 + 16 * j + l16] = f2bf(f);
      }
    }
  }
  __syncthreads();

  // ---- layer 3: [16x256] @ w3 -> y ----
  for (int nh = 0; nh < 2; ++nh) {
    const int n0 = nh * 128;
    v8f acc[8];
#pragma unroll
    for (int j = 0; j < 8; ++j) acc[j] = vzero;
    for (int kc = 0; kc < 8; ++kc) {
      const int k0 = kc * 32;
      v16bf a = load_a(&h2[wave][k0], 256, l16, half);
#pragma unroll
      for (int j = 0; j < 8; ++j) {
        v16bf b = load_bt(w3T + (size_t)(n0 + 16 * j) * 256 + k0, 256, l16, half);
        acc[j] = wmma_bf16(a, b, acc[j]);
      }
    }
#pragma unroll
    for (int j = 0; j < 8; ++j) {
#pragma unroll
      for (int v = 0; v < 8; ++v) {
        float f = acc[j][v] + b3[n0 + 16 * j + l16];
        y[(row0 + v + 8 * half) * DDIM + n0 + 16 * j + l16] = f2bf(f);
      }
    }
  }
}

// ---------------------------------------------------------------------------
// S = q k^T / sqrt(256).  Wave tile: 16t x 128c.  Block 8 waves = 128t x 128c.
// ---------------------------------------------------------------------------
__global__ __launch_bounds__(256) void qk_kernel(
    const uint16_t* __restrict__ q,   // [B*NT][256]
    const uint16_t* __restrict__ k,   // [B*NC][256]  (== B^T storage for k^T)
    uint16_t* __restrict__ S)         // [B][NT][NC]
{
  const int tid  = threadIdx.x;
  const int wave = tid >> 5;
  const int lane = tid & 31;
  const int half = lane >> 4, l16 = lane & 15;
  const int b  = blockIdx.z;
  const size_t t0 = (size_t)blockIdx.x * 128 + (size_t)wave * 16;
  const int c0 = blockIdx.y * 128;
  const uint16_t* qb = q + ((size_t)b * NTGT + t0) * DDIM;
  const uint16_t* kb = k + ((size_t)b * NCTX + c0) * DDIM;
  const v8f vzero = {};
  v8f acc[8];
#pragma unroll
  for (int j = 0; j < 8; ++j) acc[j] = vzero;
  for (int kc = 0; kc < 8; ++kc) {
    const int k0 = kc * 32;
    v16bf a = load_a(qb + k0, DDIM, l16, half);
#pragma unroll
    for (int j = 0; j < 8; ++j) {
      v16bf bf = load_bt(kb + (size_t)(16 * j) * DDIM + k0, DDIM, l16, half);
      acc[j] = wmma_bf16(a, bf, acc[j]);
    }
  }
  uint16_t* Sb = S + ((size_t)b * NTGT + t0) * NCTX + c0;
#pragma unroll
  for (int j = 0; j < 8; ++j)
#pragma unroll
    for (int v = 0; v < 8; ++v)
      Sb[(size_t)(v + 8 * half) * NCTX + 16 * j + l16] = f2bf(acc[j][v] * 0.0625f);
}

// ---------------------------------------------------------------------------
// Column softmax stats (softmax over axis=1 == over t).  One thread per column.
// Overwrites S with E = exp(S - m_c); writes Z_c.
// ---------------------------------------------------------------------------
__global__ __launch_bounds__(256) void stats_kernel(uint16_t* __restrict__ S,
                                                    float* __restrict__ colsum) {
  const int b = blockIdx.y;
  const int c = blockIdx.x * 256 + threadIdx.x;
  uint16_t* col = S + (size_t)b * NTGT * NCTX + c;
  float m = -3.0e38f;
  for (int t = 0; t < NTGT; ++t) m = fmaxf(m, bf2f(col[(size_t)t * NCTX]));
  float sum = 0.0f;
  for (int t = 0; t < NTGT; ++t) {
    float e = __expf(bf2f(col[(size_t)t * NCTX]) - m);
    sum += e;
    col[(size_t)t * NCTX] = f2bf(e);
  }
  colsum[b * NCTX + c] = sum;
}

// ---------------------------------------------------------------------------
// r'T[b][d][c] = r[b][c][d] / Z_c, bf16, via 64x64 LDS tile transpose.
// ---------------------------------------------------------------------------
__global__ __launch_bounds__(256) void rscale_kernel(
    const float* __restrict__ r,       // [B][NC][256]
    const float* __restrict__ colsum,  // [B][NC]
    uint16_t* __restrict__ rT)         // [B][256][NC]
{
  __shared__ float tile[64][65];
  const int tx = threadIdx.x & 63;
  const int ty = threadIdx.x >> 6;  // 0..3
  const int c0 = blockIdx.x * 64;
  const int d0 = blockIdx.y * 64;
  const int b  = blockIdx.z;
  const float* rb = r + ((size_t)b * NCTX + c0) * DDIM + d0;
#pragma unroll
  for (int i = 0; i < 16; ++i) {
    int cl = ty + 4 * i;
    tile[cl][tx] = rb[(size_t)cl * DDIM + tx];
  }
  __syncthreads();
  float invZ = 1.0f / colsum[b * NCTX + c0 + tx];
  uint16_t* ob = rT + ((size_t)b * DDIM + d0) * NCTX + c0;
#pragma unroll
  for (int i = 0; i < 16; ++i) {
    int dl = ty + 4 * i;
    ob[(size_t)dl * NCTX + tx] = f2bf(tile[tx][dl] * invZ);
  }
}

// ---------------------------------------------------------------------------
// out = E @ r'  : pure bf16 WMMA GEMM over K = NC = 2048.
// Wave tile 16t x 128d; block 8 waves = 128t x 128d.
// ---------------------------------------------------------------------------
__global__ __launch_bounds__(256) void pv_kernel(
    const uint16_t* __restrict__ E,   // [B][NT][NC]
    const uint16_t* __restrict__ rT,  // [B][256][NC]
    float* __restrict__ out)          // [B][NT][256]
{
  const int tid  = threadIdx.x;
  const int wave = tid >> 5;
  const int lane = tid & 31;
  const int half = lane >> 4, l16 = lane & 15;
  const int b  = blockIdx.z;
  const size_t t0 = (size_t)blockIdx.x * 128 + (size_t)wave * 16;
  const int d0 = blockIdx.y * 128;
  const uint16_t* Eb = E + ((size_t)b * NTGT + t0) * NCTX;
  const uint16_t* rb = rT + ((size_t)b * DDIM + d0) * NCTX;
  const v8f vzero = {};
  v8f acc[8];
#pragma unroll
  for (int j = 0; j < 8; ++j) acc[j] = vzero;
  for (int kc = 0; kc < 64; ++kc) {
    const int k0 = kc * 32;
    __builtin_prefetch(Eb + k0 + 256, 0, 0);   // global_prefetch_b8 ahead on A
    v16bf a = load_a(Eb + k0, NCTX, l16, half);
#pragma unroll
    for (int j = 0; j < 8; ++j) {
      v16bf bf = load_bt(rb + (size_t)(16 * j) * NCTX + k0, NCTX, l16, half);
      acc[j] = wmma_bf16(a, bf, acc[j]);
    }
  }
  float* ob = out + ((size_t)b * NTGT + t0) * DDIM + d0;
#pragma unroll
  for (int j = 0; j < 8; ++j)
#pragma unroll
    for (int v = 0; v < 8; ++v)
      ob[(size_t)(v + 8 * half) * DDIM + 16 * j + l16] = acc[j][v];
}

// ---------------------------------------------------------------------------
extern "C" void kernel_launch(void* const* d_in, const int* in_sizes, int n_in,
                              void* d_out, int out_size, void* d_ws, size_t ws_size,
                              hipStream_t stream) {
  (void)in_sizes; (void)n_in; (void)out_size; (void)ws_size;
  const float* x_context = (const float*)d_in[0];
  const float* x_target  = (const float*)d_in[1];
  const float* r_i       = (const float*)d_in[2];
  const float* qw1 = (const float*)d_in[3];  const float* qb1 = (const float*)d_in[4];
  const float* qw2 = (const float*)d_in[5];  const float* qb2 = (const float*)d_in[6];
  const float* qw3 = (const float*)d_in[7];  const float* qb3 = (const float*)d_in[8];
  const float* kw1 = (const float*)d_in[9];  const float* kb1 = (const float*)d_in[10];
  const float* kw2 = (const float*)d_in[11]; const float* kb2 = (const float*)d_in[12];
  const float* kw3 = (const float*)d_in[13]; const float* kb3 = (const float*)d_in[14];
  float* out = (float*)d_out;

  uint8_t* ws = (uint8_t*)d_ws;
  size_t off = 0;
  auto alloc = [&](size_t bytes) -> uint8_t* {
    uint8_t* p = ws + off;
    off += (bytes + 255) & ~(size_t)255;
    return p;
  };
  uint16_t* wq1T = (uint16_t*)alloc((size_t)IND  * DDIM * 2);
  uint16_t* wq2T = (uint16_t*)alloc((size_t)DDIM * DDIM * 2);
  uint16_t* wq3T = (uint16_t*)alloc((size_t)DDIM * DDIM * 2);
  uint16_t* wk1T = (uint16_t*)alloc((size_t)IND  * DDIM * 2);
  uint16_t* wk2T = (uint16_t*)alloc((size_t)DDIM * DDIM * 2);
  uint16_t* wk3T = (uint16_t*)alloc((size_t)DDIM * DDIM * 2);
  uint16_t* xt   = (uint16_t*)alloc((size_t)NBATCH * NTGT * IND * 2);
  uint16_t* xc   = (uint16_t*)alloc((size_t)NBATCH * NCTX * IND * 2);
  uint16_t* qbuf = (uint16_t*)alloc((size_t)NBATCH * NTGT * DDIM * 2);
  uint16_t* kbuf = (uint16_t*)alloc((size_t)NBATCH * NCTX * DDIM * 2);
  uint16_t* rTbf = (uint16_t*)alloc((size_t)NBATCH * DDIM * NCTX * 2);
  float*    csum = (float*)   alloc((size_t)NBATCH * NCTX * 4);
  uint16_t* Sbuf = (uint16_t*)alloc((size_t)NBATCH * NTGT * NCTX * 2);

  const int T = 256;
  // weight transposes + converts
  cvt_wT_kernel<<<(IND * DDIM + T - 1) / T, T, 0, stream>>>(qw1, wq1T, IND, DDIM);
  cvt_wT_kernel<<<(DDIM * DDIM + T - 1) / T, T, 0, stream>>>(qw2, wq2T, DDIM, DDIM);
  cvt_wT_kernel<<<(DDIM * DDIM + T - 1) / T, T, 0, stream>>>(qw3, wq3T, DDIM, DDIM);
  cvt_wT_kernel<<<(IND * DDIM + T - 1) / T, T, 0, stream>>>(kw1, wk1T, IND, DDIM);
  cvt_wT_kernel<<<(DDIM * DDIM + T - 1) / T, T, 0, stream>>>(kw2, wk2T, DDIM, DDIM);
  cvt_wT_kernel<<<(DDIM * DDIM + T - 1) / T, T, 0, stream>>>(kw3, wk3T, DDIM, DDIM);
  const int nxt = NBATCH * NTGT * IND;
  const int nxc = NBATCH * NCTX * IND;
  cvt_bf16_kernel<<<(nxt + T - 1) / T, T, 0, stream>>>(x_target, xt, nxt);
  cvt_bf16_kernel<<<(nxc + T - 1) / T, T, 0, stream>>>(x_context, xc, nxc);

  // MLPs: q from targets, k from contexts (32 rows/block)
  mlp_kernel<<<(NBATCH * NTGT) / 32, 64, 0, stream>>>(xt, wq1T, qb1, wq2T, qb2,
                                                      wq3T, qb3, qbuf);
  mlp_kernel<<<(NBATCH * NCTX) / 32, 64, 0, stream>>>(xc, wk1T, kb1, wk2T, kb2,
                                                      wk3T, kb3, kbuf);

  // attention
  qk_kernel<<<dim3(NTGT / 128, NCTX / 128, NBATCH), 256, 0, stream>>>(qbuf, kbuf, Sbuf);
  stats_kernel<<<dim3(NCTX / 256, NBATCH), 256, 0, stream>>>(Sbuf, csum);
  rscale_kernel<<<dim3(NCTX / 64, DDIM / 64, NBATCH), 256, 0, stream>>>(r_i, csum, rTbf);
  pv_kernel<<<dim3(NTGT / 128, DDIM / 128, NBATCH), 256, 0, stream>>>(Sbuf, rTbf, out);
}
